// OneHeadSelfAttention_60808146977165
// MI455X (gfx1250) — matrix operand order repaired and checked by reading, not compiled, not run on hardware
//
#include <hip/hip_runtime.h>
#include <hip/hip_bf16.h>

#define D_MODEL 1024
#define SEQ     2048
#define NBATCH  4
#define BT      (NBATCH * SEQ)

typedef __attribute__((ext_vector_type(16))) __bf16       bf16x16;
typedef __attribute__((ext_vector_type(8)))  float        floatx8;
typedef __attribute__((ext_vector_type(4)))  float        floatx4;
typedef __attribute__((ext_vector_type(8)))  unsigned int uintx8;
typedef __attribute__((ext_vector_type(4)))  unsigned int uintx4;
typedef __attribute__((ext_vector_type(2)))  unsigned int uintx2;

union FragBF {
    bf16x16 h;
    uintx8  u;
};

__device__ __forceinline__ unsigned short f32_to_bf16_rne(float f) {
    unsigned int u = __float_as_uint(f);
    u += 0x7FFFu + ((u >> 16) & 1u);
    return (unsigned short)(u >> 16);
}

// Load a 16-element bf16 fragment as two 16-byte chunks (LDS -> ds_load_b128 x2)
__device__ __forceinline__ FragBF load_frag(const unsigned short* p0,
                                            const unsigned short* p1) {
    uintx4 a = *(const uintx4*)p0;
    uintx4 b = *(const uintx4*)p1;
    FragBF f;
    f.u[0] = a[0]; f.u[1] = a[1]; f.u[2] = a[2]; f.u[3] = a[3];
    f.u[4] = b[0]; f.u[5] = b[1]; f.u[6] = b[2]; f.u[7] = b[3];
    return f;
}

// CDNA5 async global->LDS DMA (GVS mode), tracked by ASYNCcnt.
// lds_addr: per-lane LDS byte address (low 32 bits of generic shared pointer)
// byte_off: per-lane byte offset added to the scalar 64-bit base.
__device__ __forceinline__ void async_copy16(unsigned int lds_addr,
                                             const unsigned short* base,
                                             unsigned int byte_off) {
    asm volatile("global_load_async_to_lds_b128 %0, %1, %2"
                 :: "v"(lds_addr), "v"(byte_off), "s"(base)
                 : "memory");
}

#define WAIT_ASYNCCNT(n) asm volatile("s_wait_asynccnt %0" :: "n"(n) : "memory")

// ---------------------------------------------------------------------------
// Kernel 1: fused Q/K/V projection.  out = x @ W^T + b  (bf16 outputs in ws)
// WG = 256 threads (8 waves), tile = 128 rows x 64 cols, K staged in 32-chunks.
// Wave (wm, wn) in 4x2 arrangement computes a 32x32 subtile for q, k and v.
// (Staging needs f32->bf16 conversion, so it must pass through VGPRs.)
// ---------------------------------------------------------------------------
__global__ __launch_bounds__(256) void qkv_proj_kernel(
    const float* __restrict__ x,
    const float* __restrict__ wq, const float* __restrict__ bq,
    const float* __restrict__ wk, const float* __restrict__ bk,
    const float* __restrict__ wv, const float* __restrict__ bv,
    unsigned short* __restrict__ qb,
    unsigned short* __restrict__ kb,
    unsigned short* __restrict__ vb)
{
    constexpr int XST = 40;  // padded LDS stride in bf16 elems (80B, 16B aligned)
    __shared__ __align__(16) unsigned short Xs[128 * XST];
    __shared__ __align__(16) unsigned short Ws[3][64 * XST];

    const float* Wm[3] = {wq, wk, wv};
    const float* Bs[3] = {bq, bk, bv};
    unsigned short* Ob[3] = {qb, kb, vb};

    const int tid   = threadIdx.x;
    const int lane  = tid & 31;
    const int wid   = tid >> 5;
    const int wm    = wid & 3;   // 0..3  (M direction, 32 rows each)
    const int wn    = wid >> 2;  // 0..1  (N direction, 32 cols each)
    const int l16   = lane & 15;
    const int lhalf = (lane < 16) ? 0 : 1;
    const int base_m = blockIdx.x * 128;
    const int base_n = blockIdx.y * 64;

    floatx8 acc[3][2][2];
#pragma unroll
    for (int w = 0; w < 3; w++)
#pragma unroll
        for (int a = 0; a < 2; a++)
#pragma unroll
            for (int b2 = 0; b2 < 2; b2++)
                acc[w][a][b2] = (floatx8)0.0f;

    for (int kc = 0; kc < D_MODEL; kc += 32) {
        // ---- stage X tile: 128 rows x 32 k, f32 -> bf16 ----
#pragma unroll
        for (int i = 0; i < 4; i++) {
            int id  = tid + 256 * i;
            int row = id >> 3;
            int c   = (id & 7) * 4;
            floatx4 xv = *(const floatx4*)(x + (size_t)(base_m + row) * D_MODEL + kc + c);
            uintx2 pk;
            pk[0] = (unsigned int)f32_to_bf16_rne(xv[0]) |
                    ((unsigned int)f32_to_bf16_rne(xv[1]) << 16);
            pk[1] = (unsigned int)f32_to_bf16_rne(xv[2]) |
                    ((unsigned int)f32_to_bf16_rne(xv[3]) << 16);
            *(uintx2*)&Xs[row * XST + c] = pk;
        }
        // ---- stage 3 weight tiles: 64 cols x 32 k each ----
#pragma unroll
        for (int w = 0; w < 3; w++) {
#pragma unroll
            for (int i = 0; i < 2; i++) {
                int id  = tid + 256 * i;
                int row = id >> 3;
                int c   = (id & 7) * 4;
                floatx4 wv4 = *(const floatx4*)(Wm[w] + (size_t)(base_n + row) * D_MODEL + kc + c);
                uintx2 pk;
                pk[0] = (unsigned int)f32_to_bf16_rne(wv4[0]) |
                        ((unsigned int)f32_to_bf16_rne(wv4[1]) << 16);
                pk[1] = (unsigned int)f32_to_bf16_rne(wv4[2]) |
                        ((unsigned int)f32_to_bf16_rne(wv4[3]) << 16);
                *(uintx2*)&Ws[w][row * XST + c] = pk;
            }
        }
        __syncthreads();

        // ---- A fragments (X rows): lanes 0-15 K {0..7,16..23}, 16-31 K {8..15,24..31}
        FragBF afr[2];
#pragma unroll
        for (int sa = 0; sa < 2; sa++) {
            int r  = wm * 32 + sa * 16 + l16;
            int c0 = lhalf * 8;
            afr[sa] = load_frag(&Xs[r * XST + c0], &Xs[r * XST + c0 + 16]);
        }
        // ---- B fragments (W rows = output cols): lanes 0-15 K {0..7,16..23}, 16-31 K {8..15,24..31}
#pragma unroll
        for (int w = 0; w < 3; w++) {
#pragma unroll
            for (int sb = 0; sb < 2; sb++) {
                int n  = wn * 32 + sb * 16 + l16;
                int o0 = lhalf * 8;
                FragBF bfr = load_frag(&Ws[w][n * XST + o0], &Ws[w][n * XST + o0 + 16]);
#pragma unroll
                for (int sa = 0; sa < 2; sa++) {
                    acc[w][sa][sb] = __builtin_amdgcn_wmma_f32_16x16x32_bf16(
                        false, afr[sa].h, false, bfr.h, (short)0,
                        acc[w][sa][sb], false, false);
                }
            }
        }
        __syncthreads();
    }

    // ---- epilogue: + bias, store bf16 (C/D layout: row = vi + 8*lhalf, col = l16)
#pragma unroll
    for (int w = 0; w < 3; w++) {
#pragma unroll
        for (int sb = 0; sb < 2; sb++) {
            int n_g = base_n + wn * 32 + sb * 16 + l16;
            float bias = Bs[w][n_g];
#pragma unroll
            for (int sa = 0; sa < 2; sa++) {
#pragma unroll
                for (int vi = 0; vi < 8; vi++) {
                    int m_g = base_m + wm * 32 + sa * 16 + vi + lhalf * 8;
                    Ob[w][(size_t)m_g * D_MODEL + n_g] =
                        f32_to_bf16_rne(acc[w][sa][sb][vi] + bias);
                }
            }
        }
    }
}

// ---------------------------------------------------------------------------
// Kernel 2: causal flash attention from bf16 Q/K/V (L2 resident), f32 out.
// Grid: (T/64 q-blocks, D/256 d-groups, B). WG = 128 threads (4 waves),
// wave w owns query rows q0+16w..+15 and 256 output columns (16 accumulators).
// Q/K tiles are staged with double-buffered GLOBAL_LOAD_ASYNC_TO_LDS_B128:
// the DMA for d-chunk i+1 overlaps the 4 WMMAs of chunk i.
// ---------------------------------------------------------------------------
__global__ __launch_bounds__(128) void flash_attn_kernel(
    const unsigned short* __restrict__ qb,
    const unsigned short* __restrict__ kb,
    const unsigned short* __restrict__ vb,
    float* __restrict__ out)
{
    constexpr int QST = 40;  // Q/K tile stride (bf16 elems, 80B, 16B aligned)
    constexpr int PST = 72;  // P / Vt stride (bf16 elems, 144B, 16B aligned)
    __shared__ __align__(16) unsigned short Qs[2][64 * QST];
    __shared__ __align__(16) unsigned short Ks[2][64 * QST];
    __shared__ __align__(16) unsigned short Ps[64 * PST];
    __shared__ __align__(16) unsigned short Vt[256 * PST];  // [d_local][key_local]

    const int tid   = threadIdx.x;
    const int lane  = tid & 31;
    const int wid   = tid >> 5;   // 0..3
    const int l16   = lane & 15;
    const int lhalf = (lane < 16) ? 0 : 1;

    const int q0  = blockIdx.x * 64;
    const int dg0 = blockIdx.y * 256;
    const int bt  = blockIdx.z;
    const size_t bbase = (size_t)bt * SEQ * D_MODEL;

    const unsigned short* qp = qb + bbase;
    const unsigned short* kp = kb + bbase;
    const unsigned short* vp = vb + bbase;

    // per-thread staging geometry: 2 chunks of 8 bf16 per 64x32 tile
    const int srow0 = tid >> 2;                 // rows for i=0
    const int scc   = (tid & 3) * 8;            // col chunk base

    floatx8 oacc[16];
#pragma unroll
    for (int i = 0; i < 16; i++) oacc[i] = (floatx8)0.0f;
    float rm[8], rl[8];
#pragma unroll
    for (int i = 0; i < 8; i++) { rm[i] = -1e30f; rl[i] = 0.0f; }

    const float scale = 0.03125f;  // 1/sqrt(1024)

    for (int kt = 0; kt <= q0; kt += 64) {
        // ================= S = (Q K^T), accumulate over d =================
        floatx8 sacc[4];
#pragma unroll
        for (int j = 0; j < 4; j++) sacc[j] = (floatx8)0.0f;

        // prologue: async-stage chunk 0 into buffer 0 (4 ops per thread/wave)
#pragma unroll
        for (int i = 0; i < 2; i++) {
            int row = srow0 + 32 * i;
            async_copy16((unsigned int)(uintptr_t)&Qs[0][row * QST + scc], qp,
                         (unsigned int)(((q0 + row) * D_MODEL + scc) * 2));
            async_copy16((unsigned int)(uintptr_t)&Ks[0][row * QST + scc], kp,
                         (unsigned int)(((kt + row) * D_MODEL + scc) * 2));
        }

        for (int ic = 0; ic < 32; ic++) {
            const int cur = ic & 1;
            const int dc  = ic * 32;
            if (ic < 31) {
                // prefetch next chunk into the other buffer, then wait for cur
                const int nxt = cur ^ 1;
                const int dcn = dc + 32;
#pragma unroll
                for (int i = 0; i < 2; i++) {
                    int row = srow0 + 32 * i;
                    async_copy16((unsigned int)(uintptr_t)&Qs[nxt][row * QST + scc], qp,
                                 (unsigned int)(((q0 + row) * D_MODEL + dcn + scc) * 2));
                    async_copy16((unsigned int)(uintptr_t)&Ks[nxt][row * QST + scc], kp,
                                 (unsigned int)(((kt + row) * D_MODEL + dcn + scc) * 2));
                }
                WAIT_ASYNCCNT(4);   // chunk `cur` landed; prefetch still in flight
            } else {
                WAIT_ASYNCCNT(0);
            }
            __syncthreads();

            int r  = wid * 16 + l16;
            int c0 = lhalf * 8;
            FragBF aq = load_frag(&Qs[cur][r * QST + c0], &Qs[cur][r * QST + c0 + 16]);
#pragma unroll
            for (int j = 0; j < 4; j++) {
                int n  = j * 16 + l16;
                int o0 = lhalf * 8;
                FragBF bk2 = load_frag(&Ks[cur][n * QST + o0], &Ks[cur][n * QST + o0 + 16]);
                sacc[j] = __builtin_amdgcn_wmma_f32_16x16x32_bf16(
                    false, aq.h, false, bk2.h, (short)0, sacc[j], false, false);
            }
            __syncthreads();
            (void)dc;
        }

        // ================= scale + causal mask + block row max ============
        float bm[8];
#pragma unroll
        for (int vi = 0; vi < 8; vi++) bm[vi] = -1e30f;
#pragma unroll
        for (int j = 0; j < 4; j++) {
            int colg = kt + j * 16 + l16;
#pragma unroll
            for (int vi = 0; vi < 8; vi++) {
                int rowg = q0 + wid * 16 + vi + lhalf * 8;
                float s = sacc[j][vi] * scale;
                if (colg > rowg) s = -1e30f;
                sacc[j][vi] = s;
                bm[vi] = fmaxf(bm[vi], s);
            }
        }
#pragma unroll
        for (int m = 1; m < 16; m <<= 1) {
#pragma unroll
            for (int vi = 0; vi < 8; vi++)
                bm[vi] = fmaxf(bm[vi], __shfl_xor(bm[vi], m, 32));
        }

        // ================= online softmax update ==========================
        float corr[8];
#pragma unroll
        for (int vi = 0; vi < 8; vi++) {
            float nm = fmaxf(rm[vi], bm[vi]);
            corr[vi] = __expf(rm[vi] - nm);
            rm[vi]   = nm;
        }
#pragma unroll
        for (int t = 0; t < 16; t++) {
#pragma unroll
            for (int vi = 0; vi < 8; vi++) oacc[t][vi] *= corr[vi];
        }

        float psum[8];
#pragma unroll
        for (int vi = 0; vi < 8; vi++) psum[vi] = 0.0f;
#pragma unroll
        for (int j = 0; j < 4; j++) {
#pragma unroll
            for (int vi = 0; vi < 8; vi++) {
                float p = __expf(sacc[j][vi] - rm[vi]);
                psum[vi] += p;
                Ps[(wid * 16 + vi + lhalf * 8) * PST + j * 16 + l16] =
                    f32_to_bf16_rne(p);
            }
        }
#pragma unroll
        for (int m = 1; m < 16; m <<= 1) {
#pragma unroll
            for (int vi = 0; vi < 8; vi++)
                psum[vi] += __shfl_xor(psum[vi], m, 32);
        }
#pragma unroll
        for (int vi = 0; vi < 8; vi++) rl[vi] = rl[vi] * corr[vi] + psum[vi];

        // ================= stage V transposed: Vt[d][key] =================
#pragma unroll
        for (int it = 0; it < 16; it++) {
            int chunk = tid + 128 * it;        // 2048 chunks of 8 bf16
            int kk = chunk >> 5;               // key row 0..63
            int dd = (chunk & 31) * 8;         // d col base 0..248
            uintx4 vv = *(const uintx4*)(vp + (size_t)(kt + kk) * D_MODEL + dg0 + dd);
#pragma unroll
            for (int e = 0; e < 4; e++) {
                unsigned int wrd = vv[e];
                Vt[(dd + 2 * e)     * PST + kk] = (unsigned short)(wrd & 0xFFFFu);
                Vt[(dd + 2 * e + 1) * PST + kk] = (unsigned short)(wrd >> 16);
            }
        }
        __syncthreads();

        // ================= O += P @ V =====================================
#pragma unroll
        for (int h = 0; h < 2; h++) {          // two 32-key halves
            int r  = wid * 16 + l16;
            int c0 = h * 32 + lhalf * 8;
            FragBF ap = load_frag(&Ps[r * PST + c0], &Ps[r * PST + c0 + 16]);
#pragma unroll
            for (int dt = 0; dt < 16; dt++) {
                int n   = dt * 16 + l16;
                int kb0 = h * 32 + lhalf * 8;
                FragBF bv2 = load_frag(&Vt[n * PST + kb0], &Vt[n * PST + kb0 + 16]);
                oacc[dt] = __builtin_amdgcn_wmma_f32_16x16x32_bf16(
                    false, ap.h, false, bv2.h, (short)0, oacc[dt], false, false);
            }
        }
        __syncthreads();
    }

    // ================= normalize and write f32 output =====================
#pragma unroll
    for (int vi = 0; vi < 8; vi++) {
        int rowg  = q0 + wid * 16 + vi + lhalf * 8;
        float inv = 1.0f / rl[vi];
        size_t ob = bbase + (size_t)rowg * D_MODEL + dg0;
#pragma unroll
        for (int dt = 0; dt < 16; dt++) {
            out[ob + dt * 16 + l16] = oacc[dt][vi] * inv;
        }
    }
}

// ---------------------------------------------------------------------------
extern "C" void kernel_launch(void* const* d_in, const int* in_sizes, int n_in,
                              void* d_out, int out_size, void* d_ws, size_t ws_size,
                              hipStream_t stream) {
    (void)in_sizes; (void)n_in; (void)out_size; (void)ws_size;
    const float* x  = (const float*)d_in[0];
    const float* wq = (const float*)d_in[1];
    const float* bq = (const float*)d_in[2];
    const float* wk = (const float*)d_in[3];
    const float* bk = (const float*)d_in[4];
    const float* wv = (const float*)d_in[5];
    const float* bv = (const float*)d_in[6];
    float* out = (float*)d_out;

    // ws layout: Q, K, V as bf16 (ushort), 16 MB each = 48 MB total
    unsigned short* qb = (unsigned short*)d_ws;
    unsigned short* kb = qb + (size_t)BT * D_MODEL;
    unsigned short* vb = kb + (size_t)BT * D_MODEL;

    dim3 g1(BT / 128, D_MODEL / 64);  // 64 x 16 workgroups
    qkv_proj_kernel<<<g1, 256, 0, stream>>>(x, wq, bq, wk, bk, wv, bv, qb, kb, vb);

    dim3 g2(SEQ / 64, D_MODEL / 256, NBATCH);  // 32 x 4 x 4 workgroups
    flash_attn_kernel<<<g2, 128, 0, stream>>>(qb, kb, vb, out);
}
